// FrequencyDomainAlpha_71416716198474
// MI455X (gfx1250) — matrix operand order repaired and checked by reading, not compiled, not run on hardware
//
#include <hip/hip_runtime.h>
#include <math.h>

// MI455X (gfx1250) fused kernel:
//   per row: 512-pt FFT (LDS, radix-2 DIF) -> top-k by |X|^2 -> tanh feats
//   per 16-row tile: f16 WMMA GEMM (f32 accum) -> LN -> GELU -> dot -> sigmoid
// Roofline: ~256MB fp32 reads / 23.3 TB/s ~= 11us floor; FFT ~3 GFLOP and the
// GEMMs ~1.3 GFLOP are negligible, so the MLP runs on v_wmma_f32_16x16x32_f16
// (tanh features in [-1,1] -> f16 inputs, f32 accumulation).
// Input tile (16x512 f32) is DMA'd to LDS with the Tensor Data Mover (6-arg
// tensor_load_to_lds builtin on this toolchain); weights are pre-packed into
// WMMA B-fragment order so fragment builds are 2x ds_load_b128 per operand.

typedef __attribute__((ext_vector_type(16))) _Float16 v16h;
typedef __attribute__((ext_vector_type(8)))  float    v8f;
typedef unsigned int v4u __attribute__((ext_vector_type(4)));
typedef int          v8i __attribute__((ext_vector_type(8)));
typedef int          v4i __attribute__((ext_vector_type(4)));

#define ROWS 16
#define DLEN 512

__device__ __forceinline__ unsigned umax32(unsigned a, unsigned b) { return a > b ? a : b; }

__device__ __forceinline__ float gelu_exact(float x) {
  return 0.5f * x * (1.0f + erff(x * 0.70710678118654752440f));
}

union FragU {
  uint4 q[2];
  v16h  h;
};

template<int KK>
__global__ __launch_bounds__(512)
void freq_alpha_kernel(const float* __restrict__ x,    // [N, 512]
                       const float* __restrict__ wA,   // [2KK, KK] row-major
                       const float* __restrict__ bA,   // [KK]
                       const float* __restrict__ gg,   // [KK]
                       const float* __restrict__ be,   // [KK]
                       const float* __restrict__ wB,   // [KK]
                       const float* __restrict__ bB,   // [1]
                       float* __restrict__ out)        // [N]
{
  constexpr int FEATW = 2 * KK;        // 128 or 64
  constexpr int HID   = KK;            // 64 or 32
  constexpr int NT    = KK / 16;       // N-tiles of 16
  constexpr int KC    = FEATW / 32;    // K-chunks of 32

  __shared__ __align__(16) float    sRe[ROWS * DLEN];
  __shared__ __align__(16) float    sIm[ROWS * DLEN];
  __shared__ float    sTwC[DLEN / 2];
  __shared__ float    sTwS[DLEN / 2];
  __shared__ __align__(16) _Float16 sFeat[ROWS * FEATW];
  // sW holds wA repacked in WMMA B-fragment order:
  //   element B[k][n], k = kc*32 + h*16 + j  ->  sW[((kc*2+h)*HID + n)*16 + j]
  __shared__ __align__(16) _Float16 sW[FEATW * HID];
  __shared__ float    sH[ROWS * HID];
  __shared__ unsigned short sSel[ROWS * KK];   // selected LDS positions

  const int tid  = threadIdx.x;
  const int lane = tid & 31;
  const int wv   = tid >> 5;                         // wave id = row-in-tile
  const long rowBase = (long)blockIdx.x * ROWS;

  // ---- input tile -> LDS ----
#if __has_builtin(__builtin_amdgcn_tensor_load_to_lds)
  if (wv == 0) {
    // Tensor DMA descriptor (D#), 2D tile: 16 rows x 512 f32, row stride 512.
    const unsigned long long ga = (unsigned long long)(x + rowBase * DLEN);
    const unsigned ldsAddr = (unsigned)(size_t)(&sRe[0]);   // flat addr[31:0] = LDS offset
    v4u g0;
    g0[0] = 1u;                                   // count=1, user descriptor
    g0[1] = ldsAddr;                              // lds_addr (bytes)
    g0[2] = (unsigned)(ga & 0xFFFFFFFFull);       // global_addr[31:0]
    g0[3] = (unsigned)((ga >> 32) & 0x1FFFFFFull) // global_addr[56:32]
          | (2u << 30);                           // type=2 ("image")
    v8i g1;
    g1[0] = 0x00020000;                           // data_size=2 (4B), mask=0
    g1[1] = (int)(512u << 16);                    // tensor_dim0[15:0]=512
    g1[2] = (int)(16u << 16);                     // tensor_dim0[31:16]=0, tensor_dim1[15:0]=16
    g1[3] = (int)(512u << 16);                    // tensor_dim1[31:16]=0, tile_dim0=512
    g1[4] = 16;                                   // tile_dim1=16, tile_dim2=0
    g1[5] = 512;                                  // tensor_dim0_stride[31:0]=512
    g1[6] = 0;                                    // stride0 hi, stride1 lo
    g1[7] = 0;                                    // tensor_dim1_stride hi
    v4i gz4 = {0, 0, 0, 0};                       // groups 2/3 unused (2D)
    v8i gz8 = {0, 0, 0, 0, 0, 0, 0, 0};          // trailing group (6-arg form)
    __builtin_amdgcn_tensor_load_to_lds(g0, g1, gz4, gz4, gz8, 0);
#if __has_builtin(__builtin_amdgcn_s_wait_tensorcnt)
    __builtin_amdgcn_s_wait_tensorcnt(0);
#else
    asm volatile("s_wait_tensorcnt 0x0" ::: "memory");
#endif
  }
  {
    float4* dstI = (float4*)sIm;
    const float4 z = make_float4(0.f, 0.f, 0.f, 0.f);
    #pragma unroll
    for (int i = 0; i < (ROWS * DLEN / 4) / 512; ++i)
      dstI[tid + i * 512] = z;
  }
#else
  {
    const float4* src  = (const float4*)(x + rowBase * DLEN);
    float4*       dstR = (float4*)sRe;
    float4*       dstI = (float4*)sIm;
    const float4  z    = make_float4(0.f, 0.f, 0.f, 0.f);
    #pragma unroll
    for (int i = 0; i < (ROWS * DLEN / 4) / 512; ++i) {
      int idx = tid + i * 512;
      dstR[idx] = src[idx];
      dstI[idx] = z;
    }
  }
#endif

  // ---- weights -> f16, repacked into B-fragment order ----
  for (int i = tid; i < FEATW * HID; i += 512) {
    const int k = i / HID, n = i % HID;          // HID is a power of two
    const int kc = k >> 5, h = (k >> 4) & 1, j = k & 15;
    sW[(((kc << 1) + h) * HID + n) * 16 + j] = (_Float16)wA[i];
  }
  // ---- twiddle table ----
  if (tid < DLEN / 2) {
    float a = -6.28318530717958647692f * (float)tid / (float)DLEN;
    float s, c;
    sincosf(a, &s, &c);
    sTwC[tid] = c;
    sTwS[tid] = s;
  }
  __syncthreads();

  // ---- radix-2 DIF FFT, one wave per row (bit-reversed output order) ----
  float* re = sRe + wv * DLEN;
  float* im = sIm + wv * DLEN;
  for (int hs = 8; hs >= 0; --hs) {        // half = 1<<hs : 256..1
    const int half = 1 << hs;
    #pragma unroll
    for (int w = 0; w < 8; ++w) {
      const int t  = lane + (w << 5);      // butterfly id 0..255
      const int j  = t & (half - 1);
      const int i0 = ((t >> hs) << (hs + 1)) + j;
      const int i1 = i0 + half;
      const float ur = re[i0], ui = im[i0];
      const float vr = re[i1], vi = im[i1];
      const float br = ur - vr, bi = ui - vi;
      const int   tw = j << (8 - hs);
      const float c = sTwC[tw], s = sTwS[tw];
      re[i0] = ur + vr;  im[i0] = ui + vi;
      re[i1] = br * c - bi * s;
      im[i1] = br * s + bi * c;
    }
    __syncthreads();
  }

  // ---- per-row top-KK by |X|^2 (keys pack amp-bits + tie-break index) ----
  unsigned keys[16];
  #pragma unroll
  for (int w = 0; w < 16; ++w) {
    const int p  = lane + (w << 5);                  // bit-reversed position
    const float rr = re[p], ii = im[p];
    const unsigned ab = __float_as_uint(rr * rr + ii * ii);
    const int idx = (int)(__brev((unsigned)p) >> 23); // true FFT bin index
    keys[w] = (ab & 0xFFFFFE00u) | (unsigned)(511 - idx); // max amp, min idx
  }
  unsigned lmax = 0;
  #pragma unroll
  for (int w = 0; w < 16; ++w) lmax = umax32(lmax, keys[w]);

  for (int r = 0; r < KK; ++r) {
    unsigned gmax = lmax;
    #pragma unroll
    for (int off = 16; off >= 1; off >>= 1)
      gmax = umax32(gmax, (unsigned)__shfl_xor((int)gmax, off, 32));
    if (lane == 0) {
      const int idx = 511 - (int)(gmax & 511u);
      sSel[wv * KK + r] = (unsigned short)(__brev((unsigned)idx) >> 23);
    }
    if (lmax == gmax) {                    // unique keys -> exactly one winner
      #pragma unroll
      for (int w = 0; w < 16; ++w)
        if (keys[w] == gmax) keys[w] = 0u;
      lmax = 0;
      #pragma unroll
      for (int w = 0; w < 16; ++w) lmax = umax32(lmax, keys[w]);
    }
  }
  // wave-parallel feature gather + tanh (same wave wrote sSel -> in order)
  #pragma unroll
  for (int r0 = 0; r0 < KK; r0 += 32) {
    const int r = r0 + lane;
    const int p = (int)sSel[wv * KK + r];
    sFeat[wv * FEATW + r]      = (_Float16)tanhf(re[p]);
    sFeat[wv * FEATW + r + KK] = (_Float16)tanhf(im[p]);
  }
  __syncthreads();

  // ---- MLP GEMM: [16 x FEATW] @ [FEATW x HID] via v_wmma_f32_16x16x32_f16 ----
  if (wv == 0) {
    const int m     = lane & 15;
    const int koff  = (lane < 16) ? 0 : 16;  // A: K k0..15 vs k0+16..31 (matches B)
    const int hblk  = (lane < 16) ? 0 : 1;   // B: K rows k0..15 vs k0+16..31
    const int nl    = lane & 15;
    const int mbase = (lane < 16) ? 0 : 8;
    #pragma unroll
    for (int nt = 0; nt < NT; ++nt) {
      v8f acc = {0.f, 0.f, 0.f, 0.f, 0.f, 0.f, 0.f, 0.f};
      const int n = nt * 16 + nl;
      #pragma unroll
      for (int kc = 0; kc < KC; ++kc) {
        const int k0 = kc * 32;
        FragU ua, ub;
        const _Float16* ap = &sFeat[m * FEATW + k0];
        ua.q[0] = *(const uint4*)(ap + koff);        // K = k0+koff   .. +7
        ua.q[1] = *(const uint4*)(ap + koff + 8);    // K = k0+koff+8 .. +15
        const _Float16* bp = &sW[(((kc << 1) + hblk) * HID + n) * 16];
        ub.q[0] = *(const uint4*)(bp);               // j = 0..7
        ub.q[1] = *(const uint4*)(bp + 8);           // j = 8..15
        acc = __builtin_amdgcn_wmma_f32_16x16x32_f16(
            /*neg_a=*/false, ua.h, /*neg_b=*/false, ub.h,
            /*c_mod=*/(short)0, acc, /*reuse_a=*/false, /*reuse_b=*/false);
      }
      #pragma unroll
      for (int j = 0; j < 8; ++j)
        sH[(mbase + j) * HID + n] = acc[j];
    }
  }
  __syncthreads();

  // ---- bias + LayerNorm(HID) + GELU + dot(wB) + sigmoid, one wave per row ----
  {
    const int c0 = lane;
    float h0 = sH[wv * HID + c0] + bA[c0];
    float h1 = 0.0f;
    if constexpr (KK == 64) h1 = sH[wv * HID + c0 + 32] + bA[c0 + 32];
    float s = h0 + h1;
    #pragma unroll
    for (int off = 16; off >= 1; off >>= 1) s += __shfl_xor(s, off, 32);
    const float mu = s * (1.0f / (float)HID);
    const float d0 = h0 - mu;
    const float d1 = (KK == 64) ? (h1 - mu) : 0.0f;
    float q = d0 * d0 + d1 * d1;
    #pragma unroll
    for (int off = 16; off >= 1; off >>= 1) q += __shfl_xor(q, off, 32);
    const float rstd = rsqrtf(q * (1.0f / (float)HID) + 1e-5f);
    float acc = gelu_exact(d0 * rstd * gg[c0] + be[c0]) * wB[c0];
    if constexpr (KK == 64)
      acc += gelu_exact(d1 * rstd * gg[c0 + 32] + be[c0 + 32]) * wB[c0 + 32];
    #pragma unroll
    for (int off = 16; off >= 1; off >>= 1) acc += __shfl_xor(acc, off, 32);
    if (lane == 0)
      out[rowBase + wv] = 1.0f / (1.0f + expf(-(acc + bB[0])));
  }
}

extern "C" void kernel_launch(void* const* d_in, const int* in_sizes, int n_in,
                              void* d_out, int out_size, void* d_ws, size_t ws_size,
                              hipStream_t stream) {
  (void)in_sizes; (void)n_in; (void)d_ws; (void)ws_size; (void)out_size;
  const float* res1 = (const float*)d_in[0];
  const float* res2 = (const float*)d_in[1];
  const float* w1a  = (const float*)d_in[2];
  const float* b1a  = (const float*)d_in[3];
  const float* g1   = (const float*)d_in[4];
  const float* be1  = (const float*)d_in[5];
  const float* w1b  = (const float*)d_in[6];
  const float* b1b  = (const float*)d_in[7];
  const float* w2a  = (const float*)d_in[8];
  const float* b2a  = (const float*)d_in[9];
  const float* g2   = (const float*)d_in[10];
  const float* be2  = (const float*)d_in[11];
  const float* w2b  = (const float*)d_in[12];
  const float* b2b  = (const float*)d_in[13];
  float* out = (float*)d_out;

  const int N = 65536;
  dim3 grid(N / ROWS), block(512);
  freq_alpha_kernel<64><<<grid, block, 0, stream>>>(res1, w1a, b1a, g1, be1, w1b, b1b, out);
  freq_alpha_kernel<32><<<grid, block, 0, stream>>>(res2, w2a, b2a, g2, be2, w2b, b2b, out + N);
}
